// MHSA_54855322304622
// MI455X (gfx1250) — matrix operand order repaired and checked by reading, not compiled, not run on hardware
//
#include <hip/hip_runtime.h>

typedef __attribute__((ext_vector_type(16))) _Float16 v16h;
typedef __attribute__((ext_vector_type(8)))  _Float16 v8h;
typedef __attribute__((ext_vector_type(8)))  float    v8f;
typedef __attribute__((ext_vector_type(4)))  int      v4i;
typedef __attribute__((ext_vector_type(2)))  int      v2i;

#define NB     1024   // batch
#define DMODEL 512
#define NH     8
#define HD     64     // head dim
#define NTOK   81
#define NPAD   96     // padded token count (6 x 16)

// ---- CDNA5 async global->LDS path (guarded; falls back cleanly) ----------
#if defined(__HIP_DEVICE_COMPILE__) && __has_builtin(__builtin_amdgcn_global_load_async_to_lds_b128)
#define HAVE_ASYNC 1
#else
#define HAVE_ASYNC 0
#endif
#if defined(__HIP_DEVICE_COMPILE__) && __has_builtin(__builtin_amdgcn_global_load_async_to_lds_b64)
#define HAVE_ASYNC64 1
#else
#define HAVE_ASYNC64 0
#endif

#if HAVE_ASYNC
__device__ __forceinline__ void async_cp16(const _Float16* gsrc, _Float16* ldst) {
    __builtin_amdgcn_global_load_async_to_lds_b128((v4i*)gsrc, (v4i*)ldst, 0, 0);
}
#if HAVE_ASYNC64
__device__ __forceinline__ void async_cp8(const _Float16* gsrc, _Float16* ldst) {
    __builtin_amdgcn_global_load_async_to_lds_b64((v2i*)gsrc, (v2i*)ldst, 0, 0);
}
#endif
__device__ __forceinline__ void async_wait0() {
#if __has_builtin(__builtin_amdgcn_s_wait_asynccnt)
    __builtin_amdgcn_s_wait_asynccnt(0);
#else
    asm volatile("s_wait_asynccnt 0x0" ::: "memory");
#endif
}
#endif

// ---- WMMA fragment helpers (16x16x32 f16 -> f32) -------------------------
__device__ __forceinline__ v16h frag_a(const _Float16* base, int stride, int row, int kbase, int lane) {
    int m = lane & 15, hf = lane >> 4;
    const _Float16* p = base + (size_t)(row + m) * stride + kbase + hf * 16;
    v8h lo = *(const v8h*)p;
    v8h hi = *(const v8h*)(p + 8);
    return __builtin_shufflevector(lo, hi, 0,1,2,3,4,5,6,7,8,9,10,11,12,13,14,15);
}
__device__ __forceinline__ v16h frag_b(const _Float16* base, int stride, int col, int kbase, int lane) {
    int m = lane & 15, hf = lane >> 4;
    return *(const v16h*)(base + (size_t)(col + m) * stride + kbase + hf * 16);
}
__device__ __forceinline__ v8f wmma16(v16h a, v16h b, v8f c) {
    return __builtin_amdgcn_wmma_f32_16x16x32_f16(false, a, false, b, (short)0, c, false, false);
}

// ---- prep: W fp32 -> f16 -------------------------------------------------
__global__ __launch_bounds__(256) void prep_w_kernel(const float* __restrict__ wsrc,
                                                     _Float16* __restrict__ wh) {
    int i = blockIdx.x * 256 + threadIdx.x;
    if (i < 3 * DMODEL * DMODEL) wh[i] = (_Float16)wsrc[i];
}

// ---- prep: rT[h][n][d] = rel_h[h,d,j] + rel_w[h,d,i]  (n=i*9+j; pad->0) ---
__global__ __launch_bounds__(256) void prep_r_kernel(const float* __restrict__ rh,
                                                     const float* __restrict__ rw,
                                                     _Float16* __restrict__ rT) {
    int idx = blockIdx.x * 256 + threadIdx.x;
    if (idx >= NH * NPAD * HD) return;
    int h = idx / (NPAD * HD);
    int rem = idx - h * NPAD * HD;
    int n = rem / HD;
    int d = rem - n * HD;
    float v = 0.f;
    if (n < NTOK) {
        int i = n / 9, j = n - i * 9;
        v = rh[(h * HD + d) * 9 + j] + rw[(h * HD + d) * 9 + i];
    }
    rT[idx] = (_Float16)v;
}

// ---- prep: transpose + pad + convert x -> xh[b][n (96)][c (512)] f16 ------
__global__ __launch_bounds__(256) void prep_x_kernel(const float* __restrict__ x,
                                                     _Float16* __restrict__ xh) {
    const int b = blockIdx.x;
    const int t = threadIdx.x;
    const float* xb = x + (size_t)b * DMODEL * NTOK;
    _Float16* xo = xh + (size_t)b * NPAD * DMODEL;
    for (int n = 0; n < NPAD; ++n) {
        for (int c = t; c < DMODEL; c += 256) {
            float v = (n < NTOK) ? xb[(size_t)c * NTOK + n] : 0.f;
            xo[(size_t)n * DMODEL + c] = (_Float16)v;   // coalesced writes
        }
    }
}

// ---- Pass B: QKV projection GEMM, double-buffered LDS --------------------
// C[1536,96] = W[1536,512] x Xh^T + bias, per batch. grid=(12, NB), 256 thr = 8 waves.
__global__ __launch_bounds__(256) void qkv_gemm_kernel(
    const _Float16* __restrict__ xh, const _Float16* __restrict__ wh,
    const float* __restrict__ bias, const _Float16* __restrict__ rT,
    _Float16* __restrict__ qT, _Float16* __restrict__ kpr, _Float16* __restrict__ vw)
{
    __shared__ alignas(32) _Float16 Ws[2][128 * 32];   // [row][k]
    __shared__ alignas(32) _Float16 Xs[2][NPAD * 32];  // [n][k]

    const int t = threadIdx.x;
    const int lane = t & 31, w = t >> 5;
    const int rb = blockIdx.x * 128;
    const int bb = blockIdx.y;
    const _Float16* wsrc = wh + (size_t)rb * DMODEL;         // [128][512]
    const _Float16* xsrc = xh + (size_t)bb * NPAD * DMODEL;  // [96][512]

    v8f acc[6];
    const v8f vzero = {0.f,0.f,0.f,0.f,0.f,0.f,0.f,0.f};
#pragma unroll
    for (int ct = 0; ct < 6; ++ct) acc[ct] = vzero;

#if HAVE_ASYNC
    // stage one K-chunk (32 wide) into buffer `buf` via async DMA
    auto stage = [&](int kc, int buf) {
        const int k0 = kc * 32;
        int i = t;                                   // W: 512 transfers of 8 halfs
        async_cp16(wsrc + (size_t)(i >> 2) * DMODEL + k0 + (i & 3) * 8,
                   &Ws[buf][(i >> 2) * 32 + (i & 3) * 8]);
        i = t + 256;
        async_cp16(wsrc + (size_t)(i >> 2) * DMODEL + k0 + (i & 3) * 8,
                   &Ws[buf][(i >> 2) * 32 + (i & 3) * 8]);
#if HAVE_ASYNC64
        // X: 768 transfers of 4 halfs -> exactly 3 per thread, no divergence
#pragma unroll
        for (int e = 0; e < 3; ++e) {
            int j = t + e * 256;
            async_cp8(xsrc + (size_t)(j >> 3) * DMODEL + k0 + (j & 7) * 4,
                      &Xs[buf][(j >> 3) * 32 + (j & 7) * 4]);
        }
#else
        int j = t;                                   // X: 384 transfers of 8 halfs
        async_cp16(xsrc + (size_t)(j >> 2) * DMODEL + k0 + (j & 3) * 8,
                   &Xs[buf][(j >> 2) * 32 + (j & 3) * 8]);
        if (t < 128) {
            j = t + 256;
            async_cp16(xsrc + (size_t)(j >> 2) * DMODEL + k0 + (j & 3) * 8,
                       &Xs[buf][(j >> 2) * 32 + (j & 3) * 8]);
        }
#endif
    };
    stage(0, 0);
    async_wait0();
#else
    const int wrow = t >> 1, wpart = (t & 1) * 16;   // 16 halfs per thread
    const bool xact = t < 192;
    {
        v16h wr = *(const v16h*)(wsrc + (size_t)wrow * DMODEL + wpart);
        *(v16h*)(&Ws[0][wrow * 32 + wpart]) = wr;
        if (xact) {
            v16h xr = *(const v16h*)(xsrc + (size_t)wrow * DMODEL + wpart);
            *(v16h*)(&Xs[0][wrow * 32 + wpart]) = xr;
        }
    }
#endif
    __syncthreads();

    for (int kc = 0; kc < 16; ++kc) {
        const int cur = kc & 1, nxt = cur ^ 1;
        const bool more = (kc + 1) < 16;
#if HAVE_ASYNC
        if (more) stage(kc + 1, nxt);
#else
        v16h wr, xr;
        if (more) {
            const int k0 = (kc + 1) * 32;
            wr = *(const v16h*)(wsrc + (size_t)wrow * DMODEL + k0 + wpart);
            if (xact) xr = *(const v16h*)(xsrc + (size_t)wrow * DMODEL + k0 + wpart);
        }
#endif
        if (more) __builtin_prefetch(wsrc + (size_t)(t >> 1) * DMODEL + (kc + 2) * 32, 0, 1);

        // gather all fragments first, then run the WMMA chain
        v16h a = frag_a(Ws[cur], 32, w * 16, 0, lane);
        v16h bfr[6];
#pragma unroll
        for (int ct = 0; ct < 6; ++ct) bfr[ct] = frag_b(Xs[cur], 32, ct * 16, 0, lane);
#pragma unroll
        for (int ct = 0; ct < 6; ++ct) acc[ct] = wmma16(a, bfr[ct], acc[ct]);

#if HAVE_ASYNC
        async_wait0();
#else
        if (more) {
            *(v16h*)(&Ws[nxt][wrow * 32 + wpart]) = wr;
            if (xact) *(v16h*)(&Xs[nxt][wrow * 32 + wpart]) = xr;
        }
#endif
        __syncthreads();
    }

    // Epilogue: row o = obase+i, col n = ct*16 + (lane&15)
    const int hf = lane >> 4;
    const int col = lane & 15;
    const int obase = rb + w * 16 + hf * 8;
    const int sec = obase >> 9;        // 0=q 1=k 2=v
    const int rem = obase & 511;
    const int h = rem >> 6;
    const int dbase = rem & 63;

    float bval[8];
#pragma unroll
    for (int i = 0; i < 8; ++i) bval[i] = bias[obase + i];

#pragma unroll
    for (int ct = 0; ct < 6; ++ct) {
        const int n = ct * 16 + col;
        if (sec == 0) {
            v8h pk;
#pragma unroll
            for (int i = 0; i < 8; ++i) pk[i] = (_Float16)(acc[ct][i] + bval[i]);
            *(v8h*)(qT + (((size_t)bb * NH + h) * NPAD + n) * HD + dbase) = pk;
        } else if (sec == 1) {
            v8h r = *(const v8h*)(rT + ((size_t)h * NPAD + n) * HD + dbase);
            v8h pk;
#pragma unroll
            for (int i = 0; i < 8; ++i) pk[i] = (_Float16)(acc[ct][i] + bval[i] + (float)r[i]);
            *(v8h*)(kpr + (((size_t)bb * NH + h) * NPAD + n) * HD + dbase) = pk;
        } else {
#pragma unroll
            for (int i = 0; i < 8; ++i)
                vw[(((size_t)bb * NH + h) * HD + dbase + i) * NPAD + n] =
                    (_Float16)(acc[ct][i] + bval[i]);
        }
    }
}

// ---- Pass C: attention per (b,h) -----------------------------------------
__global__ __launch_bounds__(192) void attn_kernel(
    const _Float16* __restrict__ qT, const _Float16* __restrict__ kpr,
    const _Float16* __restrict__ vw, float* __restrict__ out)
{
    __shared__ alignas(32) _Float16 q_s[NPAD * HD];   // [n][d]
    __shared__ alignas(32) _Float16 k_s[NPAD * HD];   // [m][d]
    __shared__ alignas(32) _Float16 v_s[HD * NPAD];   // [d][m]
    __shared__ alignas(32) _Float16 p_s[NPAD * NPAD]; // [n][m]

    const int t = threadIdx.x;
    const int lane = t & 31, w = t >> 5;
    const int blk = blockIdx.x;                 // b*8 + h
    const size_t base = (size_t)blk * NPAD * HD;

#if HAVE_ASYNC
    for (int i = t; i < (NPAD * HD) / 8; i += 192) {
        async_cp16(qT  + base + (size_t)i * 8, q_s + i * 8);
        async_cp16(kpr + base + (size_t)i * 8, k_s + i * 8);
        async_cp16(vw  + base + (size_t)i * 8, v_s + i * 8);
    }
    async_wait0();
#else
    {
        const v8h* qsrc = (const v8h*)(qT + base);
        const v8h* ksrc = (const v8h*)(kpr + base);
        const v8h* vsrc = (const v8h*)(vw + base);
        v8h* qdst = (v8h*)q_s; v8h* kdst = (v8h*)k_s; v8h* vdst = (v8h*)v_s;
        for (int i = t; i < (NPAD * HD) / 8; i += 192) {
            qdst[i] = qsrc[i]; kdst[i] = ksrc[i]; vdst[i] = vsrc[i];
        }
    }
#endif
    __syncthreads();

    // score GEMM: strip rows [w*16, w*16+16), K=64
    v8f sacc[6];
    const v8f vzero = {0.f,0.f,0.f,0.f,0.f,0.f,0.f,0.f};
#pragma unroll
    for (int ct = 0; ct < 6; ++ct) sacc[ct] = vzero;
#pragma unroll
    for (int kc = 0; kc < 2; ++kc) {
        v16h a = frag_a(q_s, HD, w * 16, kc * 32, lane);
        v16h bfr[6];
#pragma unroll
        for (int ct = 0; ct < 6; ++ct) bfr[ct] = frag_b(k_s, HD, ct * 16, kc * 32, lane);
#pragma unroll
        for (int ct = 0; ct < 6; ++ct) sacc[ct] = wmma16(a, bfr[ct], sacc[ct]);
    }

    // in-register masked softmax
    const int col = lane & 15, hf = lane >> 4;
#pragma unroll
    for (int i = 0; i < 8; ++i) {
        const int row = w * 16 + i + hf * 8;
        float mv = -1e30f;
#pragma unroll
        for (int ct = 0; ct < 6; ++ct) {
            int c = ct * 16 + col;
            float vv = (c < NTOK) ? sacc[ct][i] : -1e30f;
            sacc[ct][i] = vv;
            mv = fmaxf(mv, vv);
        }
#pragma unroll
        for (int off = 1; off < 16; off <<= 1)
            mv = fmaxf(mv, __shfl_xor(mv, off, 32));
        float s = 0.f;
#pragma unroll
        for (int ct = 0; ct < 6; ++ct) {
            float e = __expf(sacc[ct][i] - mv);
            sacc[ct][i] = e;
            s += e;
        }
#pragma unroll
        for (int off = 1; off < 16; off <<= 1)
            s += __shfl_xor(s, off, 32);
        const float inv = 1.f / s;
#pragma unroll
        for (int ct = 0; ct < 6; ++ct)
            p_s[row * NPAD + ct * 16 + col] = (_Float16)(sacc[ct][i] * inv);
    }
    __syncthreads();

    // PV GEMM: O[64,96] = V[64x96] x P^T; 24 tiles over 6 waves
    const int bb = blk >> 3, h = blk & 7;
    float* ob = out + ((size_t)bb * DMODEL + h * HD) * NTOK;
#pragma unroll
    for (int j = 0; j < 4; ++j) {
        const int tIdx = w * 4 + j;
        const int td = tIdx / 6, tn = tIdx - 6 * td;
        v8f oacc = vzero;
#pragma unroll
        for (int kc = 0; kc < 3; ++kc) {
            v16h a = frag_a(v_s, NPAD, td * 16, kc * 32, lane);
            oacc = wmma16(a, frag_b(p_s, NPAD, tn * 16, kc * 32, lane), oacc);
        }
        const int n = tn * 16 + col;
        if (n < NTOK) {
#pragma unroll
            for (int i = 0; i < 8; ++i) {
                const int d = td * 16 + i + hf * 8;
                ob[(size_t)d * NTOK + n] = oacc[i];
            }
        }
    }
}

// --------------------------------------------------------------------------
extern "C" void kernel_launch(void* const* d_in, const int* in_sizes, int n_in,
                              void* d_out, int out_size, void* d_ws, size_t ws_size,
                              hipStream_t stream) {
    (void)in_sizes; (void)n_in; (void)out_size; (void)ws_size;
    const float* x     = (const float*)d_in[0];
    const float* qkv_w = (const float*)d_in[1];
    const float* qkv_b = (const float*)d_in[2];
    const float* rel_h = (const float*)d_in[3];
    const float* rel_w = (const float*)d_in[4];
    float* out = (float*)d_out;

    _Float16* ws = (_Float16*)d_ws;
    size_t off = 0;
    _Float16* wh  = ws + off; off += (size_t)3 * DMODEL * DMODEL;      // 1.5 MB
    _Float16* rT  = ws + off; off += (size_t)NH * NPAD * HD;           // 96 KB
    _Float16* xh  = ws + off; off += (size_t)NB * NPAD * DMODEL;       // 100 MB
    _Float16* qT  = ws + off; off += (size_t)NB * NH * NPAD * HD;      // 100 MB
    _Float16* kpr = ws + off; off += (size_t)NB * NH * NPAD * HD;      // 100 MB
    _Float16* vw  = ws + off;                                          // 100 MB

    prep_w_kernel<<<(3 * DMODEL * DMODEL + 255) / 256, 256, 0, stream>>>(qkv_w, wh);
    prep_r_kernel<<<(NH * NPAD * HD + 255) / 256, 256, 0, stream>>>(rel_h, rel_w, rT);
    prep_x_kernel<<<NB, 256, 0, stream>>>(x, xh);
    qkv_gemm_kernel<<<dim3(12, NB), 256, 0, stream>>>(xh, wh, qkv_b, rT, qT, kpr, vw);
    attn_kernel<<<NB * NH, 192, 0, stream>>>(qT, kpr, vw, out);
}
